// MultiHeadSelfAttention_57449482552037
// MI455X (gfx1250) — matrix operand order repaired and checked by reading, not compiled, not run on hardware
//
#include <hip/hip_runtime.h>
#include <hip/hip_bf16.h>
#include <cstdint>

// ---------------- CDNA5 WMMA types ----------------
typedef __attribute__((ext_vector_type(16))) __bf16 bf16x16;
typedef __attribute__((ext_vector_type(8)))  float  f32x8;
typedef int v4i __attribute__((vector_size(16)));

union Frag {
    bf16x16  v;
    uint32_t w[8];
};

__device__ __forceinline__ f32x8 wmma_bf16(const Frag& a, const Frag& b, f32x8 c) {
    return __builtin_amdgcn_wmma_f32_16x16x32_bf16(false, a.v, false, b.v,
                                                   (short)0, c, false, false);
}

// ---------------- bf16 helpers ----------------
__device__ __forceinline__ uint16_t f2bf(float f) {
    uint32_t u = __builtin_bit_cast(uint32_t, f);
    u += 0x7FFFu + ((u >> 16) & 1u);   // round to nearest even
    return (uint16_t)(u >> 16);
}
__device__ __forceinline__ uint32_t pack2(float a, float b) {
    return (uint32_t)f2bf(a) | ((uint32_t)f2bf(b) << 16);
}
__device__ __forceinline__ float bf2f(uint16_t h) {
    return __builtin_bit_cast(float, (uint32_t)h << 16);
}

// Problem constants
constexpr int Bb  = 4;
constexpr int Ss  = 2048;
constexpr int Dd  = 1024;
constexpr int Hh  = 16;
constexpr int DK  = 64;

// A-fragment word selector: VGPR i -> word {0,1,2,3,8,9,10,11}[i] (+4 for lanes>=16)
__device__ __forceinline__ int awsel(int i) { return (i < 4) ? i : 4 + i; }

// ---------------- CDNA5 async global->LDS copy (16B per lane) ----------------
#if __has_builtin(__builtin_amdgcn_global_load_async_to_lds_b128)
#define ASYNC_LDS 1
#endif

__device__ __forceinline__ void copy16_to_lds(const uint32_t* g, uint32_t* l) {
#if defined(ASYNC_LDS)
    typedef __attribute__((address_space(1))) v4i* gp_t;
    typedef __attribute__((address_space(3))) v4i* lp_t;
    __builtin_amdgcn_global_load_async_to_lds_b128(
        (gp_t)(uintptr_t)g, (lp_t)(uint32_t)(uintptr_t)l, 0, 0);
#else
    ((uint4*)l)[0] = ((const uint4*)g)[0];
#endif
}

__device__ __forceinline__ void wait_async_all() {
#if defined(ASYNC_LDS)
#if __has_builtin(__builtin_amdgcn_s_wait_asynccnt)
    __builtin_amdgcn_s_wait_asynccnt(0);
#else
    asm volatile("s_wait_asynccnt 0x0" ::: "memory");
#endif
#endif
}

// =====================================================================
// Prep 1: convert x (f32) to packed-bf16 words [M][K/2]
// =====================================================================
__global__ __launch_bounds__(256) void cvt_x_kernel(const float* __restrict__ src,
                                                    uint32_t* __restrict__ dst)
{
    int id = blockIdx.x * 256 + threadIdx.x;          // one word = 2 elements
    dst[id] = pack2(src[2 * (size_t)id], src[2 * (size_t)id + 1]);
}

// =====================================================================
// Prep 2: swizzle weights f32 [K][N] -> k-pair packed bf16 words [K/2][N]
//         word(kp, n) = { W[2kp][n], W[2kp+1][n] }   (B-fragment native)
// =====================================================================
__global__ __launch_bounds__(256) void swz_w_kernel(const float* __restrict__ W0,
                                                    const float* __restrict__ W1,
                                                    const float* __restrict__ W2,
                                                    const float* __restrict__ W3,
                                                    uint32_t* __restrict__ out)
{
    int mtx = blockIdx.y;
    const float* W = (mtx == 0) ? W0 : (mtx == 1) ? W1 : (mtx == 2) ? W2 : W3;
    uint32_t* dst = out + (size_t)mtx * (Dd / 2) * Dd;
    int id = blockIdx.x * 256 + threadIdx.x;          // 524288 words per matrix
    int kp = id >> 10, n = id & (Dd - 1);
    dst[id] = pack2(W[(size_t)(2 * kp) * Dd + n], W[(size_t)(2 * kp + 1) * Dd + n]);
}

// =====================================================================
// Double-buffered bf16 WMMA GEMM:  Out = A[M,K] x B[K,N]
//   A: packed bf16 words [M][K/2];  B: pre-swizzled words [K/2][N]
//   Tiles stream into LDS via async global->LDS b128 copies (ASYNCcnt),
//   next tile issued while WMMAs consume the current one.
//   HEAD_SCATTER=true : bf16 output scattered head-major [b,h,s,64]
//   HEAD_SCATTER=false: f32 output row-major [M,N]
// Block tile 128x128x32, 256 threads = 8 waves (4x2), each wave 32x64.
// =====================================================================
template <bool HEAD_SCATTER>
__global__ __launch_bounds__(256) void gemm_wmma(
    const uint32_t* __restrict__ Aw,
    const uint32_t* __restrict__ Bw,
    void* __restrict__ Out,
    int M, int N, int K)
{
    constexpr int BM = 128, BN = 128, BK = 32;
    constexpr int AR = 20;    // padded A row words (16 used), 16B-aligned rows
    constexpr int BR = 132;   // padded B row words (128 used), 16B-aligned rows
    __shared__ uint32_t As[2][BM][AR];
    __shared__ uint32_t Bs[2][BK / 2][BR];

    const int tid  = threadIdx.x;
    const int lane = tid & 31;
    const int wave = tid >> 5;
    const int wm   = wave & 3;
    const int wn   = wave >> 2;
    const int lrow = lane & 15;
    const int lhi  = lane >> 4;
    const int row0 = blockIdx.x * BM;
    const int col0 = blockIdx.y * BN;
    const int K2   = K >> 1;

    auto issue_tile = [&](int t, int buf) {
        const int kw0 = t * (BK / 2);
        // A tile: 128 rows x 16 words = 512 x b128 chunks
#pragma unroll
        for (int i = 0; i < 2; ++i) {
            int idx = tid + i * 256;
            int r = idx >> 2, c4 = idx & 3;
            copy16_to_lds(Aw + (size_t)(row0 + r) * K2 + kw0 + c4 * 4,
                          &As[buf][r][c4 * 4]);
        }
        // B tile: 16 kp-rows x 128 words = 512 x b128 chunks
#pragma unroll
        for (int i = 0; i < 2; ++i) {
            int idx = tid + i * 256;
            int kp = idx >> 5, n4 = idx & 31;
            copy16_to_lds(Bw + (size_t)(kw0 + kp) * N + col0 + n4 * 4,
                          &Bs[buf][kp][n4 * 4]);
        }
    };

    f32x8 acc[2][4];
    const f32x8 zero = {0.f, 0.f, 0.f, 0.f, 0.f, 0.f, 0.f, 0.f};
#pragma unroll
    for (int im = 0; im < 2; ++im)
#pragma unroll
        for (int nt = 0; nt < 4; ++nt) acc[im][nt] = zero;

    const int T = K / BK;
    issue_tile(0, 0);

    for (int t = 0; t < T; ++t) {
        wait_async_all();       // own tile-t copies have landed
        __syncthreads();        // everyone's tile-t copies landed; buf^1 free
        if (t + 1 < T) issue_tile(t + 1, (t + 1) & 1);

        const int buf = t & 1;
        Frag a[2];
#pragma unroll
        for (int im = 0; im < 2; ++im) {
            int r = wm * 32 + im * 16 + lrow;
#pragma unroll
            for (int i = 0; i < 8; ++i) a[im].w[i] = As[buf][r][lhi * 8 + i];
        }
#pragma unroll
        for (int nt = 0; nt < 4; ++nt) {
            Frag b;
            int n = wn * 64 + nt * 16 + lrow;
#pragma unroll
            for (int i = 0; i < 8; ++i) b.w[i] = Bs[buf][lhi * 8 + i][n];
            acc[0][nt] = wmma_bf16(a[0], b, acc[0][nt]);
            acc[1][nt] = wmma_bf16(a[1], b, acc[1][nt]);
        }
    }

    // ---- epilogue ----
#pragma unroll
    for (int im = 0; im < 2; ++im) {
#pragma unroll
        for (int nt = 0; nt < 4; ++nt) {
#pragma unroll
            for (int v = 0; v < 8; ++v) {
                int row = row0 + wm * 32 + im * 16 + lhi * 8 + v;
                int col = col0 + wn * 64 + nt * 16 + lrow;
                float val = acc[im][nt][v];
                if constexpr (HEAD_SCATTER) {
                    int bb = row >> 11, s = row & (Ss - 1);
                    int h = col >> 6, j = col & (DK - 1);
                    ((uint16_t*)Out)[(((size_t)(bb * Hh + h) * Ss) + s) * DK + j] = f2bf(val);
                } else {
                    ((float*)Out)[(size_t)row * N + col] = val;
                }
            }
        }
    }
}

// =====================================================================
// RoPE: in-place rotation of adjacent bf16 pairs in head-major Q / K.
// =====================================================================
__global__ __launch_bounds__(256) void rope_kernel(uint32_t* __restrict__ Qh,
                                                   uint32_t* __restrict__ Kh,
                                                   const int* __restrict__ pos)
{
    uint32_t* buf = (blockIdx.y == 0) ? Qh : Kh;
    int id  = blockIdx.x * 256 + threadIdx.x;   // 64 * 2048 * 32 pairs
    int bh  = id >> 16;
    int rem = id & 65535;
    int s   = rem >> 5;
    int m   = rem & 31;                         // pair index (j = 2m)
    int b   = bh >> 4;

    float p   = (float)pos[b * Ss + s];
    float inv = __powf(10000.0f, -(float)m * (1.0f / 32.0f));
    float sn, cs;
    __sincosf(p * inv, &sn, &cs);

    size_t off = ((size_t)bh * Ss + s) * (DK / 2) + m;
    uint32_t w = buf[off];
    float x1 = bf2f((uint16_t)(w & 0xFFFFu));
    float x2 = bf2f((uint16_t)(w >> 16));
    buf[off] = pack2(x1 * cs - x2 * sn, x1 * sn + x2 * cs);
}

// =====================================================================
// Flash attention: one block per (b,h) x 128 query rows; 8 waves x 16 rows.
// =====================================================================
__global__ __launch_bounds__(256) void attn_kernel(
    const uint32_t* __restrict__ Qh, const uint32_t* __restrict__ Kh,
    const uint16_t* __restrict__ Vh, uint16_t* __restrict__ attn)
{
    constexpr int KC = 64;
    __shared__ uint32_t Kl[KC][DK / 2 + 1];
    __shared__ uint32_t Vt[KC / 2][DK + 1];
    __shared__ uint16_t Pl[8][16][KC + 4];

    const int tid  = threadIdx.x;
    const int lane = tid & 31;
    const int wave = tid >> 5;
    const int lrow = lane & 15;
    const int lhi  = lane >> 4;
    const int bh   = blockIdx.y;
    const int q0   = blockIdx.x * 128;
    const size_t baseW = (size_t)bh * Ss * (DK / 2);

    Frag aq[2];
    {
        const uint32_t* qrow = Qh + baseW + (size_t)(q0 + wave * 16 + lrow) * (DK / 2);
#pragma unroll
        for (int f = 0; f < 2; ++f)
#pragma unroll
            for (int i = 0; i < 8; ++i)
                aq[f].w[i] = qrow[f * 16 + lhi * 8 + i];
    }

    float mi[8], li[8];
    f32x8 o[4];
    const f32x8 zero = {0.f, 0.f, 0.f, 0.f, 0.f, 0.f, 0.f, 0.f};
#pragma unroll
    for (int v = 0; v < 8; ++v) { mi[v] = -1e30f; li[v] = 0.f; }
#pragma unroll
    for (int nt = 0; nt < 4; ++nt) o[nt] = zero;

    const uint16_t* vb = Vh + (size_t)bh * Ss * DK;

    for (int kt = 0; kt < Ss; kt += KC) {
        __syncthreads();
#pragma unroll
        for (int i = 0; i < 8; ++i) {
            int idx = tid + i * 256;
            int r = idx >> 5, wp = idx & 31;
            Kl[r][wp] = Kh[baseW + (size_t)(kt + r) * (DK / 2) + wp];
        }
#pragma unroll
        for (int i = 0; i < 8; ++i) {
            int idx = tid + i * 256;
            int d = idx & 63, kp = idx >> 6;
            uint32_t lo = vb[(size_t)(kt + 2 * kp) * DK + d];
            uint32_t up = vb[(size_t)(kt + 2 * kp + 1) * DK + d];
            Vt[kp][d] = lo | (up << 16);
        }
        __syncthreads();

        // prefetch next K/V tile into caches while computing this one
        if (kt + KC < Ss && tid < KC) {
            __builtin_prefetch(&Kh[baseW + (size_t)(kt + KC + tid) * (DK / 2)], 0, 0);
            __builtin_prefetch(&vb[(size_t)(kt + KC + tid) * DK], 0, 0);
        }

        // ---- S = Q K^T ----
        f32x8 sacc[4];
#pragma unroll
        for (int nt = 0; nt < 4; ++nt) sacc[nt] = zero;
#pragma unroll
        for (int f = 0; f < 2; ++f) {
#pragma unroll
            for (int nt = 0; nt < 4; ++nt) {
                Frag bk;
                int key = nt * 16 + lrow;
#pragma unroll
                for (int i = 0; i < 8; ++i) bk.w[i] = Kl[key][f * 16 + lhi * 8 + i];
                sacc[nt] = wmma_bf16(aq[f], bk, sacc[nt]);
            }
        }

        // ---- online softmax (scale 0.125) ----
#pragma unroll
        for (int v = 0; v < 8; ++v) {
            float rm = fmaxf(fmaxf(sacc[0][v], sacc[1][v]),
                             fmaxf(sacc[2][v], sacc[3][v]));
            rm = fmaxf(rm, __shfl_xor(rm, 1, 32));
            rm = fmaxf(rm, __shfl_xor(rm, 2, 32));
            rm = fmaxf(rm, __shfl_xor(rm, 4, 32));
            rm = fmaxf(rm, __shfl_xor(rm, 8, 32));
            float mn = fmaxf(mi[v], rm);
            float alpha = __expf((mi[v] - mn) * 0.125f);
            float rs = 0.f;
#pragma unroll
            for (int nt = 0; nt < 4; ++nt) {
                float pe = __expf((sacc[nt][v] - mn) * 0.125f);
                Pl[wave][lhi * 8 + v][nt * 16 + lrow] = f2bf(pe);
                rs += pe;
            }
            rs += __shfl_xor(rs, 1, 32);
            rs += __shfl_xor(rs, 2, 32);
            rs += __shfl_xor(rs, 4, 32);
            rs += __shfl_xor(rs, 8, 32);
            li[v] = li[v] * alpha + rs;
            mi[v] = mn;
#pragma unroll
            for (int nt = 0; nt < 4; ++nt) o[nt][v] *= alpha;
        }

        // ---- O += P V ----
        Frag ap[2];
#pragma unroll
        for (int f = 0; f < 2; ++f)
#pragma unroll
            for (int i = 0; i < 8; ++i) {
                int word = f * 16 + lhi * 8 + i;
                ap[f].w[i] = *(const uint32_t*)&Pl[wave][lrow][2 * word];
            }
#pragma unroll
        for (int f = 0; f < 2; ++f) {
#pragma unroll
            for (int nt = 0; nt < 4; ++nt) {
                Frag bv;
                int d = nt * 16 + lrow;
#pragma unroll
                for (int i = 0; i < 8; ++i) bv.w[i] = Vt[f * 16 + lhi * 8 + i][d];
                o[nt] = wmma_bf16(ap[f], bv, o[nt]);
            }
        }
    }

    int b = bh >> 4, h = bh & 15;
#pragma unroll
    for (int nt = 0; nt < 4; ++nt) {
#pragma unroll
        for (int v = 0; v < 8; ++v) {
            int srow = q0 + wave * 16 + lhi * 8 + v;
            int col  = h * DK + nt * 16 + lrow;
            float val = o[nt][v] / li[v];
            attn[(size_t)(b * Ss + srow) * Dd + col] = f2bf(val);
        }
    }
}

// =====================================================================
extern "C" void kernel_launch(void* const* d_in, const int* in_sizes, int n_in,
                              void* d_out, int out_size, void* d_ws, size_t ws_size,
                              hipStream_t stream)
{
    const float* x   = (const float*)d_in[0];
    const int*   pos = (const int*)d_in[1];
    const float* Wq  = (const float*)d_in[2];
    const float* Wk  = (const float*)d_in[3];
    const float* Wv  = (const float*)d_in[4];
    const float* Wo  = (const float*)d_in[5];

    char* ws = (char*)d_ws;
    const size_t HSZ = (size_t)Bb * Hh * Ss * DK * sizeof(uint16_t);     // 16 MB
    uint16_t* Qh   = (uint16_t*)(ws + 0 * HSZ);
    uint16_t* Kh   = (uint16_t*)(ws + 1 * HSZ);
    uint16_t* Vh   = (uint16_t*)(ws + 2 * HSZ);
    uint16_t* attn = (uint16_t*)(ws + 3 * HSZ);
    uint32_t* Xw   = (uint32_t*)(ws + 4 * HSZ);                          // 16 MB
    uint32_t* Wsw  = (uint32_t*)(ws + 5 * HSZ);                          // 8 MB
    const size_t WSZ = (size_t)(Dd / 2) * Dd;     // words per swizzled matrix

    const int M = Bb * Ss, N = Dd, K = Dd;

    // Prep: bf16-pack activations, swizzle all 4 weight matrices
    cvt_x_kernel<<<(M * (K / 2)) / 256, 256, 0, stream>>>(x, Xw);
    swz_w_kernel<<<dim3(((Dd / 2) * Dd) / 256, 4), 256, 0, stream>>>(Wq, Wk, Wv, Wo, Wsw);

    dim3 gg(M / 128, N / 128);
    // QKV projections -> head-major bf16
    gemm_wmma<true><<<gg, 256, 0, stream>>>(Xw, Wsw + 0 * WSZ, Qh, M, N, K);
    gemm_wmma<true><<<gg, 256, 0, stream>>>(Xw, Wsw + 1 * WSZ, Kh, M, N, K);
    gemm_wmma<true><<<gg, 256, 0, stream>>>(Xw, Wsw + 2 * WSZ, Vh, M, N, K);

    // RoPE in-place on Q and K
    rope_kernel<<<dim3((Bb * Hh * Ss * (DK / 2)) / 256, 2), 256, 0, stream>>>(
        (uint32_t*)Qh, (uint32_t*)Kh, pos);

    // Flash attention
    attn_kernel<<<dim3(Ss / 128, Bb * Hh), 256, 0, stream>>>(
        (const uint32_t*)Qh, (const uint32_t*)Kh, Vh, attn);

    // Output projection -> f32 d_out
    gemm_wmma<false><<<gg, 256, 0, stream>>>((const uint32_t*)attn, Wsw + 3 * WSZ,
                                             (float*)d_out, M, N, K);
}
